// SeqTagging_64888365908579
// MI455X (gfx1250) — compile-verified, matrix operand order checked by reading, not run on hardware
//
#include <hip/hip_runtime.h>
#include <stdint.h>

typedef _Float16 half_t;
typedef _Float16 v8h  __attribute__((ext_vector_type(8)));
typedef _Float16 v16h __attribute__((ext_vector_type(16)));
typedef float    v8f  __attribute__((ext_vector_type(8)));

#define M_ROWS 16384
#define SEQ_L  256
#define NEGV   (-100000.0f)
// C = 11, START = 9, STOP = 10

static __device__ __forceinline__ float sigf(float x) { return 1.0f / (1.0f + expf(-x)); }

static __device__ __forceinline__ v8f wmma_f16(v16h a, v16h b, v8f c) {
  // D = A(16x32 f16) * B(32x16 f16) + C(16x16 f32)
  return __builtin_amdgcn_wmma_f32_16x16x32_f16(false, a, false, b, (short)0, c, false, false);
}

// gfx1250 async Global->LDS copy (ASYNCcnt-tracked), 16 bytes per lane.
static __device__ __forceinline__ void async_copy_b128(uint32_t lds_byte_off, const void* gaddr) {
  asm volatile("global_load_async_to_lds_b128 %0, %1, off"
               :: "v"(lds_byte_off), "v"(gaddr)
               : "memory");
}
static __device__ __forceinline__ void wait_async0() {
  asm volatile("s_wait_asynccnt 0x0" ::: "memory");
}

// ---------------------------------------------------------------- utilities

__global__ void k_zero_u32(uint32_t* p, int n) {
  for (int i = blockIdx.x * blockDim.x + threadIdx.x; i < n; i += gridDim.x * blockDim.x)
    p[i] = 0u;
}

// gather embeddings -> f16 activation buffer (stride 1024, cols 300..1023 stay zero)
__global__ void k_embed(const int* __restrict__ tokens, const float* __restrict__ emb,
                        half_t* __restrict__ x16) {
  int row = blockIdx.x;
  int d = threadIdx.x;
  if (d < 300) {
    int tok = tokens[row];
    x16[(size_t)row * 1024 + d] = (half_t)emb[(size_t)tok * 300 + d];
  }
}

// Pack a (N x K) f32 matrix (element (n,k) at src[n*ldn + k*ldk]) into WMMA
// B-fragment order: frag(n_tile,c) -> 32 lanes x 16 halfs contiguous.
// B 32x16 f16 layout: lane: n = lane&15, khalf = lane>>4;
// element e: j=e>>1, sub=e&1; k = (j<4 ? 2j : 16+2(j-4)) + 8*khalf + sub.
__global__ void k_pack_b(const float* __restrict__ src, half_t* __restrict__ dst,
                         int N, int K, int chunks, int n_tiles, int ldn, int ldk,
                         int dst_tile_stride) {
  int total = n_tiles * chunks * 512;
  for (int id = blockIdx.x * blockDim.x + threadIdx.x; id < total;
       id += gridDim.x * blockDim.x) {
    int e = id & 15;
    int lane = (id >> 4) & 31;
    int frag = id >> 9;
    int c = frag % chunks;
    int n_tile = frag / chunks;
    int n = n_tile * 16 + (lane & 15);
    int khalf = lane >> 4;
    int j = e >> 1, sub = e & 1;
    int kk = (j < 4 ? 2 * j : 16 + 2 * (j - 4)) + khalf * 8 + sub;
    int k = c * 32 + kk;
    float v = (n < N && k < K) ? src[(size_t)n * ldn + (size_t)k * ldk] : 0.0f;
    dst[((size_t)(n_tile * dst_tile_stride + c)) * 512 + lane * 16 + e] = (half_t)v;
  }
}

// -------------------------------------------------- conv1d as implicit GEMM
// out[b,l,o] = relu(bias[o] + sum_{kw,i} w[o,i,kw] * x[b, l+kw-2, i])
// A rows = flattened (b,l), padded channels 300->320 (10 chunks of K=32).
__global__ __launch_bounds__(128) void k_conv_gemm(const half_t* __restrict__ A,
                                                   const half_t* __restrict__ Bp,
                                                   const float* __restrict__ bias,
                                                   half_t* __restrict__ out) {
  int w = threadIdx.x >> 5, lane = threadIdx.x & 31;
  int n_tile = blockIdx.y * 4 + w;
  if (n_tile >= 19) return;
  int m0 = blockIdx.x * 16;
  int row = m0 + (lane & 15);
  int khalf = lane >> 4;
  int l = row & 255;
  v8f acc = {};
  const v8h z8 = {};
  for (int kw = 0; kw < 5; ++kw) {
    int srcl = l + kw - 2;
    bool valid = (unsigned)srcl < 256u;
    int srow = (row & ~255) | (srcl & 255);
    const half_t* pa = A + (size_t)srow * 1024 + khalf * 8;
    const half_t* pb = Bp + ((size_t)(n_tile * 5 + kw) * 10) * 512 + lane * 16;
#pragma unroll
    for (int c = 0; c < 10; ++c) {
      union { v16h v; v8h h[2]; } a;
      if (valid) {
        a.h[0] = *(const v8h*)(pa + c * 32);
        a.h[1] = *(const v8h*)(pa + c * 32 + 16);
      } else {
        a.h[0] = z8; a.h[1] = z8;
      }
      v16h bf = *(const v16h*)(pb + (size_t)c * 512);
      acc = wmma_f16(a.v, bf, acc);
    }
  }
  int col = n_tile * 16 + (lane & 15);
  if (col < 300) {
    float bb = bias[col];
#pragma unroll
    for (int r = 0; r < 8; ++r) {
      int mrow = m0 + r + 8 * khalf;
      float v = acc[r] + bb;
      v = v > 0.0f ? v : 0.0f;
      out[(size_t)mrow * 1024 + col] = (half_t)v;
    }
  }
}

// ---------------------------------------- xg = x @ Wih^T + b  (M x 2048 GEMM)
// Block = 4 waves sharing one 16-row A tile, staged through LDS with the
// gfx1250 async Global->LDS path; B-fragments stream from global (prepacked).
__global__ __launch_bounds__(128) void k_gemm_xg(const half_t* __restrict__ A,
                                                 const half_t* __restrict__ Bp,
                                                 const float* __restrict__ bias,
                                                 float* __restrict__ out, int chunks) {
  __shared__ __align__(16) half_t atile[16 * 128];  // 4 KB stage, row stride 128 halfs
  int tid = threadIdx.x;
  int w = tid >> 5, lane = tid & 31;
  int n_tile = blockIdx.y * 4 + w;  // 0..127
  int m0 = blockIdx.x * 16;
  int khalf = lane >> 4;
  v8f acc = {};
  const half_t* pb = Bp + (size_t)n_tile * chunks * 512 + lane * 16;
  uint32_t lds0 = (uint32_t)(uintptr_t)&atile[0];

  for (int g0 = 0; g0 < chunks; g0 += 4) {
    int gs = chunks - g0;
    if (gs > 4) gs = 4;
    int units = gs * 64;  // 16-byte units this stage (16 rows x gs*32 halfs)
    for (int u = tid; u < units; u += 128) {
      int r = u / (gs * 4);
      int seg = u % (gs * 4);
      const half_t* gp = A + (size_t)(m0 + r) * 1024 + g0 * 32 + seg * 8;
      async_copy_b128(lds0 + (uint32_t)(r * 256 + seg * 16), gp);
    }
    // prefetch next group's B fragments while the async copy lands
    __builtin_prefetch((const void*)(pb + (size_t)(g0 + gs) * 512), 0, 1);
    wait_async0();
    __syncthreads();
#pragma unroll 4
    for (int cc = 0; cc < gs; ++cc) {
      union { v16h v; v8h h[2]; } a;
      const half_t* pa = &atile[(lane & 15) * 128 + cc * 32 + khalf * 8];
      a.h[0] = *(const v8h*)pa;
      a.h[1] = *(const v8h*)(pa + 16);
      v16h bf = *(const v16h*)(pb + (size_t)(g0 + cc) * 512);
      acc = wmma_f16(a.v, bf, acc);
    }
    __syncthreads();  // all waves done reading before the tile is overwritten
  }

  int col = n_tile * 16 + (lane & 15);
  float bb = bias[col];
#pragma unroll
  for (int r = 0; r < 8; ++r) {
    int mrow = m0 + r + 8 * khalf;
    out[(size_t)mrow * 2048 + col] = acc[r] + bb;
  }
}

// ------------------------------------------------- LSTM recurrence (1 WG, 32 waves)
// gates(64x2048) = xg[:,idx_t,:] + h(64x512) @ Whh^T, WMMA per step;
// h lives in LDS (f16), c lives in VGPRs. wave w owns h-cols [16w,16w+16).
__global__ __launch_bounds__(1024) void k_lstm_rec(const float* __restrict__ xg,
                                                   const half_t* __restrict__ WhhP,
                                                   const int* __restrict__ lengths,
                                                   half_t* __restrict__ out16,
                                                   float* __restrict__ out32, int dir) {
  __shared__ half_t h_lds[64 * 512];  // 64 KB
  int tid = threadIdx.x;
  int w = tid >> 5, lane = tid & 31;
  int khalf = lane >> 4, nloc = lane & 15;
  for (int i = tid; i < 64 * 512; i += 1024) h_lds[i] = (half_t)0.0f;
  __syncthreads();

  float c_reg[4][8];
  int len_r[4][8];
#pragma unroll
  for (int q = 0; q < 4; ++q)
#pragma unroll
    for (int r = 0; r < 8; ++r) {
      c_reg[q][r] = 0.0f;
      len_r[q][r] = lengths[q * 16 + r + 8 * khalf];
    }
  int hcol = w * 16 + nloc;

  for (int t = 0; t < SEQ_L; ++t) {
    for (int q = 0; q < 4; ++q) {
      v8f acc0 = {}, acc1 = {}, acc2 = {}, acc3 = {};
      int arow = q * 16 + nloc;
      const half_t* pa = &h_lds[arow * 512 + khalf * 8];
      const half_t* pb = WhhP + (size_t)lane * 16;
#pragma unroll 4
      for (int c = 0; c < 16; ++c) {
        union { v16h v; v8h h[2]; } a;
        a.h[0] = *(const v8h*)(pa + c * 32);
        a.h[1] = *(const v8h*)(pa + c * 32 + 16);
        acc0 = wmma_f16(a.v, *(const v16h*)(pb + ((size_t)((0 * 32 + w) * 16 + c)) * 512), acc0);
        acc1 = wmma_f16(a.v, *(const v16h*)(pb + ((size_t)((1 * 32 + w) * 16 + c)) * 512), acc1);
        acc2 = wmma_f16(a.v, *(const v16h*)(pb + ((size_t)((2 * 32 + w) * 16 + c)) * 512), acc2);
        acc3 = wmma_f16(a.v, *(const v16h*)(pb + ((size_t)((3 * 32 + w) * 16 + c)) * 512), acc3);
      }
      __syncthreads();  // reads of rows [16q,16q+16) done before writes below
#pragma unroll
      for (int r = 0; r < 8; ++r) {
        int b = q * 16 + r + 8 * khalf;
        int len = len_r[q][r];
        int idx = (dir == 0) ? t : (t < len ? len - 1 - t : t);
        bool m = t < len;
        size_t base = ((size_t)(b * SEQ_L + idx)) * 2048 + hcol;
        float gi = acc0[r] + xg[base];
        float gf = acc1[r] + xg[base + 512];
        float gg = acc2[r] + xg[base + 1024];
        float go = acc3[r] + xg[base + 1536];
        float cn = sigf(gf) * c_reg[q][r] + sigf(gi) * tanhf(gg);
        float hn = sigf(go) * tanhf(cn);
        float ov = m ? hn : 0.0f;
        if (m) {
          c_reg[q][r] = cn;
          h_lds[b * 512 + hcol] = (half_t)hn;
        }
        size_t ob = ((size_t)(b * SEQ_L + idx)) * 1024 + (size_t)dir * 512 + hcol;
        out32[ob] = ov;
        out16[ob] = (half_t)ov;
      }
      __syncthreads();
    }
  }
}

// -------------------------------------------------------- FC: emis = x @ fc_w^T + b
__global__ void k_fc(const float* __restrict__ x, const float* __restrict__ w,
                     const float* __restrict__ bias, float* __restrict__ emis) {
  int id = blockIdx.x * blockDim.x + threadIdx.x;
  int row = id >> 4, c = id & 15;
  if (row >= M_ROWS || c >= 11) return;
  const float4* xr = (const float4*)(x + (size_t)row * 1024);
  const float4* wr = (const float4*)(w + (size_t)c * 1024);
  float s = bias[c];
  for (int k = 0; k < 256; ++k) {
    float4 a = xr[k], b = wr[k];
    s += a.x * b.x + a.y * b.y + a.z * b.z + a.w * b.w;
  }
  emis[(size_t)row * 11 + c] = s;
}

// ------------------------------------------------------------------ CRF loss
__global__ void k_crf_loss(const float* __restrict__ emis, const int* __restrict__ tags,
                           const int* __restrict__ lengths, const float* __restrict__ T,
                           float* __restrict__ out) {
  __shared__ float red[64];
  int b = threadIdx.x;
  float fs[11];
#pragma unroll
  for (int i = 0; i < 11; ++i) fs[i] = NEGV;
  fs[9] = 0.0f;  // START
  int len = lengths[b];
  for (int t = 0; t < len; ++t) {
    const float* x = emis + ((size_t)(b * SEQ_L + t)) * 11;
    float ns[11];
    for (int i = 0; i < 11; ++i) {
      float mx = fs[0] + T[i * 11 + 0];
      for (int j = 1; j < 11; ++j) {
        float v = fs[j] + T[i * 11 + j];
        mx = v > mx ? v : mx;
      }
      float s = 0.0f;
      for (int j = 0; j < 11; ++j) s += expf(fs[j] + T[i * 11 + j] - mx);
      ns[i] = mx + logf(s) + x[i];
    }
    for (int i = 0; i < 11; ++i) fs[i] = ns[i];
  }
  float mx = fs[0] + T[10 * 11 + 0];
  for (int c = 1; c < 11; ++c) {
    float v = fs[c] + T[10 * 11 + c];
    mx = v > mx ? v : mx;
  }
  float s = 0.0f;
  for (int c = 0; c < 11; ++c) s += expf(fs[c] + T[10 * 11 + c] - mx);
  float fscr = mx + logf(s);
  float gs = 0.0f;
  int prev = 9;
  for (int t = 0; t < len; ++t) {
    int tg = tags[b * SEQ_L + t];
    gs += emis[((size_t)(b * SEQ_L + t)) * 11 + tg] + T[tg * 11 + prev];
    prev = tg;
  }
  gs += T[10 * 11 + prev];
  red[b] = fscr - gs;
  __syncthreads();
  if (b == 0) {
    float acc = 0.0f;
    for (int i = 0; i < 64; ++i) acc += red[i];
    out[0] = acc / 64.0f;
  }
}

// --------------------------------------------------------------- CRF Viterbi
__global__ void k_crf_viterbi(const float* __restrict__ emis, const int* __restrict__ lengths,
                              const float* __restrict__ T, int* __restrict__ bptrs,
                              float* __restrict__ out) {
  int b = threadIdx.x;
  float ms[11];
#pragma unroll
  for (int i = 0; i < 11; ++i) ms[i] = NEGV;
  ms[9] = 0.0f;  // START
  int len = lengths[b];
  for (int t = 0; t < len; ++t) {
    const float* x = emis + ((size_t)(b * SEQ_L + t)) * 11;
    float ns[11];
    for (int i = 0; i < 11; ++i) {
      float best = ms[0] + T[i * 11 + 0];
      int bj = 0;
      for (int j = 1; j < 11; ++j) {
        float v = ms[j] + T[i * 11 + j];
        if (v > best) { best = v; bj = j; }
      }
      ns[i] = best + x[i];
      bptrs[(size_t)t * 704 + b * 11 + i] = bj;
    }
    for (int i = 0; i < 11; ++i) ms[i] = ns[i];
  }
  for (int i = 0; i < 11; ++i) ms[i] += T[10 * 11 + i];
  for (int i = 0; i < 11; ++i) out[1 + b * 11 + i] = ms[i];
  int cur = 0;
  float best = ms[0];
  for (int i = 1; i < 11; ++i)
    if (ms[i] > best) { best = ms[i]; cur = i; }
  for (int t = SEQ_L - 1; t >= 0; --t) {
    if (t < len) {
      out[705 + (size_t)b * SEQ_L + t] = (float)cur;
      cur = bptrs[(size_t)t * 704 + b * 11 + cur];
    } else {
      out[705 + (size_t)b * SEQ_L + t] = 0.0f;
    }
  }
}

// ------------------------------------------------------------------ launcher

extern "C" void kernel_launch(void* const* d_in, const int* in_sizes, int n_in,
                              void* d_out, int out_size, void* d_ws, size_t ws_size,
                              hipStream_t stream) {
  (void)in_sizes; (void)n_in; (void)out_size; (void)ws_size;
  const int* tokens = (const int*)d_in[0];
  const int* lengths = (const int*)d_in[1];
  // d_in[2] = mk (unused: recomputed from lengths)
  const int* tags = (const int*)d_in[3];
  const float* emb = (const float*)d_in[4];
  const float* conv_w = (const float*)d_in[5];
  const float* conv_b = (const float*)d_in[6];
  const float* Wih[2][2] = {{(const float*)d_in[7], (const float*)d_in[10]},
                            {(const float*)d_in[13], (const float*)d_in[16]}};
  const float* Whh[2][2] = {{(const float*)d_in[8], (const float*)d_in[11]},
                            {(const float*)d_in[14], (const float*)d_in[17]}};
  const float* bv[2][2] = {{(const float*)d_in[9], (const float*)d_in[12]},
                           {(const float*)d_in[15], (const float*)d_in[18]}};
  const float* fc_w = (const float*)d_in[19];
  const float* fc_b = (const float*)d_in[20];
  const float* Tm = (const float*)d_in[21];
  float* outp = (float*)d_out;

  char* ws = (char*)d_ws;
  half_t* XA = (half_t*)(ws + 0);                  //  33,554,432 B  f16 act ping
  half_t* XB = (half_t*)(ws + 33554432);           //  33,554,432 B  f16 act pong
  float* XG = (float*)(ws + 67108864);             // 134,217,728 B  gates buffer
  float* CC = (float*)(ws + 201326592);            //  67,108,864 B  f32 concat
  float* EMIS = (float*)(ws + 268435456);          //     720,896 B
  int* BPTR = (int*)(ws + 269156352);              //     720,896 B
  half_t* CONVP = (half_t*)(ws + 269877248);       //   1,945,600 B packed conv w
  half_t* WIHP = (half_t*)(ws + 271822848);        //  11,010,048 B packed Wih
  half_t* WHHP = (half_t*)(ws + 282832896);        //   8,388,608 B packed Whh

  dim3 b256(256);
  // zero f16 activation buffers (pad columns must be 0 for WMMA K-chunks)
  hipLaunchKernelGGL(k_zero_u32, dim3(4096), b256, 0, stream, (uint32_t*)XA, M_ROWS * 512);
  hipLaunchKernelGGL(k_zero_u32, dim3(4096), b256, 0, stream, (uint32_t*)XB, M_ROWS * 512);

  // ---- weight packing into WMMA B-fragment order
  // Wih: l0 (K=300, 10 chunks), l1 (K=1024, 32 chunks); offsets in halfs
  const size_t wih_off[2][2] = {{0, 655360}, {1310720, 3407872}};
  const int wih_chunks[2] = {10, 32};
  const int wih_K[2] = {300, 1024};
  for (int l = 0; l < 2; ++l)
    for (int d = 0; d < 2; ++d) {
      int total = 128 * wih_chunks[l] * 512;
      hipLaunchKernelGGL(k_pack_b, dim3((total + 255) / 256), b256, 0, stream,
                         Wih[l][d], WIHP + wih_off[l][d], 2048, wih_K[l], wih_chunks[l],
                         128, wih_K[l], 1, wih_chunks[l]);
    }
  // Whh: K=512, 16 chunks, 1,048,576 halfs per (layer,dir)
  for (int l = 0; l < 2; ++l)
    for (int d = 0; d < 2; ++d) {
      int total = 128 * 16 * 512;
      hipLaunchKernelGGL(k_pack_b, dim3((total + 255) / 256), b256, 0, stream,
                         Whh[l][d], WHHP + (size_t)(l * 2 + d) * 1048576, 2048, 512, 16,
                         128, 512, 1, 16);
    }
  // conv weights: per (layer, kw): N=300 (19 tiles), K=300 (10 chunks), kw-strided src
  for (int l = 0; l < 2; ++l)
    for (int kw = 0; kw < 5; ++kw) {
      int total = 19 * 10 * 512;
      hipLaunchKernelGGL(k_pack_b, dim3((total + 255) / 256), b256, 0, stream,
                         conv_w + (size_t)l * 450000 + kw, CONVP + (size_t)l * 486400 + kw * 5120,
                         300, 300, 10, 19, 1500, 5, 50);
    }

  // ---- embedding gather
  hipLaunchKernelGGL(k_embed, dim3(M_ROWS), dim3(320), 0, stream, tokens, emb, XA);

  // ---- conv stack (implicit GEMM, WMMA)
  hipLaunchKernelGGL(k_conv_gemm, dim3(1024, 5), dim3(128), 0, stream, XA, CONVP, conv_b, XB);
  hipLaunchKernelGGL(k_conv_gemm, dim3(1024, 5), dim3(128), 0, stream, XB, CONVP + 486400,
                     conv_b + 300, XA);

  // ---- BiLSTM layer 0 (input = XA, output -> XB f16 + CC f32)
  hipLaunchKernelGGL(k_gemm_xg, dim3(1024, 32), dim3(128), 0, stream, XA, WIHP + wih_off[0][0],
                     bv[0][0], XG, 10);
  hipLaunchKernelGGL(k_lstm_rec, dim3(1), dim3(1024), 0, stream, XG, WHHP + 0, lengths, XB, CC, 0);
  hipLaunchKernelGGL(k_gemm_xg, dim3(1024, 32), dim3(128), 0, stream, XA, WIHP + wih_off[0][1],
                     bv[0][1], XG, 10);
  hipLaunchKernelGGL(k_lstm_rec, dim3(1), dim3(1024), 0, stream, XG, WHHP + 1048576, lengths, XB,
                     CC, 1);

  // ---- BiLSTM layer 1 (input = XB, output -> XA f16 + CC f32)
  hipLaunchKernelGGL(k_gemm_xg, dim3(1024, 32), dim3(128), 0, stream, XB, WIHP + wih_off[1][0],
                     bv[1][0], XG, 32);
  hipLaunchKernelGGL(k_lstm_rec, dim3(1), dim3(1024), 0, stream, XG, WHHP + 2097152, lengths, XA,
                     CC, 0);
  hipLaunchKernelGGL(k_gemm_xg, dim3(1024, 32), dim3(128), 0, stream, XB, WIHP + wih_off[1][1],
                     bv[1][1], XG, 32);
  hipLaunchKernelGGL(k_lstm_rec, dim3(1), dim3(1024), 0, stream, XG, WHHP + 3145728, lengths, XA,
                     CC, 1);

  // ---- FC to emissions
  hipLaunchKernelGGL(k_fc, dim3(1024), b256, 0, stream, CC, fc_w, fc_b, EMIS);

  // ---- CRF loss + Viterbi decode
  hipLaunchKernelGGL(k_crf_loss, dim3(1), dim3(64), 0, stream, EMIS, tags, lengths, Tm, outp);
  hipLaunchKernelGGL(k_crf_viterbi, dim3(1), dim3(64), 0, stream, EMIS, lengths, Tm, BPTR, outp);
}
